// VQVAE_NET_47261820125918
// MI455X (gfx1250) — matrix operand order repaired and checked by reading, not compiled, not run on hardware
//
#include <hip/hip_runtime.h>

// ---------------------------------------------------------------------------
// VQ-VAE forward for MI455X (gfx1250, wave32, WMMA).
// Memory-bound workload (~1.5 TFLOP vs ~2GB HBM traffic): bf16 WMMA GEMMs with
// fused ReLU/bias/residual epilogues, fp32 only at HBM boundaries.
// CDNA5 paths: v_wmma_f32_16x16x32_bf16, double-buffered
// global_load_async_to_lds_b128 pipelined one tap ahead (+ s_wait_asynccnt),
// global_prefetch.
// ---------------------------------------------------------------------------

typedef __attribute__((ext_vector_type(16))) __bf16 v16bf;
typedef __attribute__((ext_vector_type(8)))  float  v8f;

union Frag { uint4 q[2]; v16bf v; };
static_assert(sizeof(Frag) == 32, "frag must be 32B (16 bf16)");

__device__ __forceinline__ unsigned short f32_bf16(float f) {
  union { float f; unsigned int u; } x; x.f = f;
  unsigned int r = x.u + 0x7FFFu + ((x.u >> 16) & 1u);   // round-nearest-even
  return (unsigned short)(r >> 16);
}

#define NIMG 1024
#define DCH  512
#define HW   49
#define CC   128            // LDS channel chunk
#define SPAD (CC + 8)       // padded LDS row stride (shorts) -> 272B rows

// ---------------------------------------------------------------------------
// Implicit-GEMM conv (KH=3: pad-1 3x3; KH=1: 1x1 GEMM).
//   out[n][o][pos] = alpha * sum_{tap,c} A(in)[pos][tap,c] * W[tap][o][c]
//                    + bias[o] + res[n][o][pos]
// Block: one image x 32 output channels. 8 waves = 4 Mtiles(16 pos) x 2 Ntiles.
// Weight slabs are DMA'd LDS-direct (async) one tap ahead of the WMMAs.
// ---------------------------------------------------------------------------
template<int KH>
__global__ __launch_bounds__(256)
void conv_wmma(const float* __restrict__ in, int inLayoutNPC, int preRelu,
               const unsigned short* __restrict__ wb,   // bf16 [tap][o][c]
               const float* __restrict__ bias, float alpha,
               const float* __restrict__ res,           // nullable, NCHW
               float* __restrict__ out) {
  constexpr int TAPS  = KH * KH;
  constexpr int SROWS = (KH == 3) ? 81 : HW;             // 9x9 pad ring or 7x7
  __shared__ __align__(16) unsigned short sIn[SROWS][SPAD];
  __shared__ __align__(16) unsigned short sW[2][32][SPAD];  // double-buffered

  const int n   = blockIdx.x >> 4;
  const int ob  = (blockIdx.x & 15) * 32;
  const int tid = threadIdx.x;
  const int lane  = tid & 31;
  const int wave  = tid >> 5;
  const int mt    = wave >> 1;       // 0..3  (position tile)
  const int nt    = wave & 1;        // 0..1  (out-channel tile)
  const int lrow  = lane & 15;       // A row / B,C column within tile
  const int khalf = lane >> 4;       // lane-half (ISA 7.12.2 layouts)

  // wave-relative LDS byte addresses (flat LDS aperture: addr[31:0])
  const unsigned sWlds[2] = { (unsigned)(uintptr_t)&sW[0][0][0],
                              (unsigned)(uintptr_t)&sW[1][0][0] };

  if (KH == 3) {                     // zero ring once (interior rewritten)
    for (int i = tid; i < SROWS * SPAD; i += 256) (&sIn[0][0])[i] = 0;
  }

  int mRow = mt * 16 + lrow; if (mRow > HW - 1) mRow = HW - 1; // clamp pad rows
  const int mh = mRow / 7, mw = mRow % 7;

  v8f acc = {};

  for (int ch = 0; ch < DCH; ch += CC) {
    __syncthreads();   // sIn + both sW buffers free of readers

    // issue tap-0 weight DMA; it runs behind the input conversion below
    auto issueW = [&](int t, int buf) {
      const unsigned short* wsrc = wb + (((long)t * DCH + ob) * DCH + ch);
      for (int i = tid; i < 32 * (CC / 8); i += 256) {   // 512 x 16B chunks
        int o = i / (CC / 8), c8 = (i % (CC / 8)) * 8;
        unsigned long long ga =
            (unsigned long long)(uintptr_t)(wsrc + (long)o * DCH + c8);
        unsigned lds = sWlds[buf] + (unsigned)(o * SPAD + c8) * 2u;
        asm volatile("global_load_async_to_lds_b128 %0, %1, off"
                     :: "v"(lds), "v"(ga) : "memory");
      }
    };
    issueW(0, 0);

    // ---- stage input chunk: fp32 -> bf16, fused pre-ReLU ----
    for (int i = tid; i < HW * CC; i += 256) {
      int pos, c;
      float v;
      if (inLayoutNPC) { pos = i / CC; c = i % CC;
        v = in[((long)n * HW + pos) * DCH + ch + c];
      } else           { c = i / HW; pos = i % HW;
        v = in[((long)n * DCH + ch + c) * HW + pos];
      }
      if (preRelu) v = v > 0.f ? v : 0.f;
      int s = (KH == 3) ? ((pos / 7 + 1) * 9 + (pos % 7 + 1)) : pos;
      sIn[s][c] = f32_bf16(v);
    }
    // prefetch next channel chunk (contiguous in NCHW) into caches
    if (!inLayoutNPC && ch + CC < DCH) {
      const char* nf = (const char*)(in + ((long)n * DCH + ch + CC) * HW);
      __builtin_prefetch(nf + tid * 98, 0, 0);   // 25088B / 256 threads
    }

    for (int t = 0; t < TAPS; ++t) {
      const int cur = t & 1;
      asm volatile("s_wait_asynccnt 0" ::: "memory");  // my slab-DMA done
      __syncthreads();   // slab[cur]+sIn visible; slab[1-cur] readers drained
      if (t + 1 < TAPS) issueW(t + 1, 1 - cur);        // DMA next tap behind WMMAs

      const int sA = (KH == 3) ? ((mh + t / 3) * 9 + (mw + t % 3)) : mRow;
      const unsigned short* aRow = &sIn[sA][0];
      const unsigned short* bRow = &sW[cur][lrow][0];
      #pragma unroll
      for (int kc = 0; kc < CC; kc += 32) {
        Frag a, b;
        // A 16x32 bf16: lane half 0 -> K {0..7,16..23}, half 1 -> {8..15,24..31}
        a.q[0] = *(const uint4*)(aRow + kc + khalf * 8);
        a.q[1] = *(const uint4*)(aRow + kc + 16 + khalf * 8);
        // B 32x16 bf16: lane half 0 -> K {0..7,16..23}, half 1 -> {8..15,24..31}
        b.q[0] = *(const uint4*)(bRow + kc + khalf * 8);
        b.q[1] = *(const uint4*)(bRow + kc + 16 + khalf * 8);
        acc = __builtin_amdgcn_wmma_f32_16x16x32_bf16(
                false, a.v, false, b.v, (short)0, acc, false, false);
      }
    }
  }

  // ---- epilogue: C layout (ISA 7.12.2): VGPR r -> M = r + 8*khalf, N = lrow
  const int o = ob + nt * 16 + lrow;
  const float bo = bias ? bias[o] : 0.f;
  const long base = ((long)n * DCH + o) * HW;
  #pragma unroll
  for (int r = 0; r < 8; ++r) {
    int pos = mt * 16 + khalf * 8 + r;
    if (pos < HW) {
      float v = alpha * acc[r] + bo;
      if (res) v += res[base + pos];
      out[base + pos] = v;
    }
  }
}

// ---------------------------------------------------------------------------
// Weight repack: fp32 OIHW/IOHW (opt. spatial flip) -> bf16 [tap][o][c]
// ---------------------------------------------------------------------------
__global__ void convert_w(const float* __restrict__ src,
                          unsigned short* __restrict__ dst,
                          int KH, int flip, int iohw) {
  const int total = KH * KH * DCH * DCH;
  for (int i = blockIdx.x * blockDim.x + threadIdx.x; i < total;
       i += gridDim.x * blockDim.x) {
    int t = i / (DCH * DCH), rem = i % (DCH * DCH);
    int o = rem / DCH, c = rem % DCH;
    int kh = t / KH, kw = t % KH;
    if (flip) { kh = KH - 1 - kh; kw = KH - 1 - kw; }
    long si = iohw ? (((long)c * DCH + o) * KH + kh) * (long)KH + kw
                   : (((long)o * DCH + c) * KH + kh) * (long)KH + kw;
    dst[i] = f32_bf16(src[si]);
  }
}

// codebook row norms ||e_k||^2
__global__ void esq_kernel(const float* __restrict__ cb, float* __restrict__ esq) {
  int k = blockIdx.x * blockDim.x + threadIdx.x;
  if (k < DCH) {
    float s = 0.f;
    for (int c = 0; c < DCH; ++c) { float v = cb[(long)k * DCH + c]; s += v * v; }
    esq[k] = s;
  }
}

// argmin over codes of dist'[n][k][pos] = ||e_k||^2 - 2<ze,e_k>, gather zq NCHW
__global__ __launch_bounds__(256)
void vq_argmin_gather(const float* __restrict__ dist,
                      const float* __restrict__ codebook,
                      float* __restrict__ zq) {
  int g = blockIdx.x * blockDim.x + threadIdx.x;
  if (g >= NIMG * HW) return;
  int n = g / HW, pos = g % HW;
  const float* dn = dist + (long)n * DCH * HW + pos;
  float best = dn[0]; int bi = 0;
  for (int k = 1; k < DCH; ++k) {
    float v = dn[(long)k * HW];
    if (v < best) { best = v; bi = k; }   // strict < == first-min (jnp.argmin)
  }
  const float* e = codebook + (long)bi * DCH;
  float* zn = zq + (long)n * DCH * HW + pos;
  for (int c = 0; c < DCH; ++c) zn[(long)c * HW] = e[c];
}

// ---------------------------------------------------------------------------
static inline void conv3(const float* in, int npc, int prelu,
                         const unsigned short* w, const float* b, float alpha,
                         const float* res, float* out, hipStream_t s) {
  conv_wmma<3><<<NIMG * 16, 256, 0, s>>>(in, npc, prelu, w, b, alpha, res, out);
}
static inline void conv1(const float* in, int prelu,
                         const unsigned short* w, const float* b, float alpha,
                         const float* res, float* out, hipStream_t s) {
  conv_wmma<1><<<NIMG * 16, 256, 0, s>>>(in, 0, prelu, w, b, alpha, res, out);
}

extern "C" void kernel_launch(void* const* d_in, const int* in_sizes, int n_in,
                              void* d_out, int out_size, void* d_ws, size_t ws_size,
                              hipStream_t stream) {
  (void)in_sizes; (void)n_in; (void)out_size; (void)ws_size;
  const float* x       = (const float*)d_in[0];
  const float* enc_w1  = (const float*)d_in[1];
  const float* enc_b1  = (const float*)d_in[2];
  const float* enc_w2  = (const float*)d_in[3];
  const float* enc_b2  = (const float*)d_in[4];
  const float* enc_rw1 = (const float*)d_in[5];
  const float* enc_rb1 = (const float*)d_in[6];
  const float* enc_rw2 = (const float*)d_in[7];
  const float* enc_rb2 = (const float*)d_in[8];
  const float* codebook= (const float*)d_in[9];
  const float* dec_w1  = (const float*)d_in[10];
  const float* dec_b1  = (const float*)d_in[11];
  const float* dec_rw1 = (const float*)d_in[12];
  const float* dec_rb1 = (const float*)d_in[13];
  const float* dec_rw2 = (const float*)d_in[14];
  const float* dec_rb2 = (const float*)d_in[15];
  const float* dec_wt  = (const float*)d_in[16];
  const float* dec_bt  = (const float*)d_in[17];

  const long NA = (long)NIMG * DCH * HW;        // 25,690,112 elems per tensor
  float* xhat = (float*)d_out;
  float* ze   = xhat + NA;                      // output #2 (also pipeline tensor)
  float* zq   = xhat + 2 * NA;                  // output #3 (also decoder input)

  // ---- workspace carve-up ----
  char* ws = (char*)d_ws;
  const size_t W3 = 9UL * DCH * DCH * 2, W1 = (size_t)DCH * DCH * 2;
  unsigned short* wb_e1  = (unsigned short*)ws; ws += W3;
  unsigned short* wb_e2  = (unsigned short*)ws; ws += W3;
  unsigned short* wb_er1 = (unsigned short*)ws; ws += W3;
  unsigned short* wb_er2 = (unsigned short*)ws; ws += W1;
  unsigned short* wb_d1  = (unsigned short*)ws; ws += W3;
  unsigned short* wb_dr1 = (unsigned short*)ws; ws += W3;
  unsigned short* wb_dr2 = (unsigned short*)ws; ws += W1;
  unsigned short* wb_dt  = (unsigned short*)ws; ws += W3;
  unsigned short* wb_cb  = (unsigned short*)ws; ws += W1;
  float* esq  = (float*)ws; ws += 2048;
  float* act0 = (float*)ws; ws += NA * sizeof(float);
  float* act1 = (float*)ws; ws += NA * sizeof(float);
  float* act2 = (float*)ws; ws += NA * sizeof(float);

  // ---- weight repack to bf16 [tap][o][c] ----
  convert_w<<<2304, 256, 0, stream>>>(enc_w1,  wb_e1,  3, 0, 0);
  convert_w<<<2304, 256, 0, stream>>>(enc_w2,  wb_e2,  3, 0, 0);
  convert_w<<<2304, 256, 0, stream>>>(enc_rw1, wb_er1, 3, 0, 0);
  convert_w<<<1024, 256, 0, stream>>>(enc_rw2, wb_er2, 1, 0, 0);
  convert_w<<<2304, 256, 0, stream>>>(dec_w1,  wb_d1,  3, 0, 0);
  convert_w<<<2304, 256, 0, stream>>>(dec_rw1, wb_dr1, 3, 0, 0);
  convert_w<<<1024, 256, 0, stream>>>(dec_rw2, wb_dr2, 1, 0, 0);
  convert_w<<<2304, 256, 0, stream>>>(dec_wt,  wb_dt,  3, 1, 1);   // flip, IOHW
  convert_w<<<1024, 256, 0, stream>>>(codebook, wb_cb, 1, 0, 0);   // [k][c]
  esq_kernel<<<2, 256, 0, stream>>>(codebook, esq);

  // ---- encoder ----
  conv3(x,    1, 0, wb_e1,  enc_b1,  1.f, nullptr, act0, stream);  // NPC input
  conv3(act0, 0, 1, wb_e2,  enc_b2,  1.f, nullptr, act1, stream);  // pre-res ze
  conv3(act1, 0, 1, wb_er1, enc_rb1, 1.f, nullptr, act2, stream);
  conv1(act2,    1, wb_er2, enc_rb2, 1.f, act1,    ze,   stream);  // ze (output)

  // ---- VQ: dist' = ||e||^2 - 2 <ze, e>; argmin; gather ----
  conv1(ze, 0, wb_cb, esq, -2.f, nullptr, act0, stream);
  vq_argmin_gather<<<(NIMG * HW + 255) / 256, 256, 0, stream>>>(act0, codebook, zq);

  // ---- decoder (dec_in == zq under straight-through) ----
  conv3(zq,   0, 0, wb_d1,  dec_b1,  1.f, nullptr, act1, stream);
  conv3(act1, 0, 1, wb_dr1, dec_rb1, 1.f, nullptr, act2, stream);
  conv1(act2,    1, wb_dr2, dec_rb2, 1.f, act1,    act0, stream);
  conv3(act0, 0, 0, wb_dt,  dec_bt,  1.f, nullptr, xhat, stream);  // x_hat
}
